// T5Attention_55104430408037
// MI455X (gfx1250) — matrix operand order repaired and checked by reading, not compiled, not run on hardware
//
#include <hip/hip_runtime.h>
#include <hip/hip_bf16.h>
#include <math.h>

// ---------------------------------------------------------------------------
// Problem constants (fixed by the reference)
// ---------------------------------------------------------------------------
constexpr int BB    = 4;     // batch
constexpr int S     = 1024;  // sequence
constexpr int D     = 1024;  // model dim
constexpr int H     = 16;    // heads
constexpr int DK    = 64;    // head dim
constexpr int INNER = H * DK;        // 1024
constexpr int M_ALL = BB * S;        // 4096 rows for the big GEMMs
constexpr int NUM_BUCKETS = 32;

typedef __attribute__((ext_vector_type(16))) _Float16 v16h;
typedef __attribute__((ext_vector_type(8)))  float    v8f;

union FragAB {
    v16h v;
    uint4 q[2];
    _Float16 h[16];
};

union H4 {
    _Float16 h[4];
    uint2 u;
};

__device__ __forceinline__ v8f wmma_f16(v16h a, v16h b, v8f c) {
    // D = A(16x32) * B(32x16) + C, f32 accumulate
    return __builtin_amdgcn_wmma_f32_16x16x32_f16(
        /*neg_a=*/false, a, /*neg_b=*/false, b,
        /*c_mod=*/(short)0, c, /*reuse_a=*/false, /*reuse_b=*/false);
}

// ---------------------------------------------------------------------------
// Generic tiled GEMM: C[M,N] = A[M,K] * B[K,N]
//   A: fp32 or f16 row-major; B: fp32 row-major, converted to f16 in LDS.
//   blockIdx.z selects one of up to three (B, C) pairs so the Q/K/V
//   projections run as ONE launch (3x the blocks in flight, shared-A L2 hits).
//   OUT_MODE 0: fp32 row-major [M,N]
//   OUT_MODE 1: f16 scattered to [B,H,S,DK]  (row = b*S+s, col = h*DK+d)
// Block tile 128x128x64, 256 threads (8 waves), wave tile 64x32.
// ---------------------------------------------------------------------------
constexpr int GBM = 128, GBN = 128, GBK = 64;
constexpr int GPAD = 16; // halves; row stride 80 halves = 160B (mult of 16B)

template <typename AT, int OUT_MODE>
__global__ __launch_bounds__(256) void gemm_kernel(
    const AT* __restrict__ A,
    const float* __restrict__ B0, const float* __restrict__ B1,
    const float* __restrict__ B2,
    void* __restrict__ C0, void* __restrict__ C1, void* __restrict__ C2,
    int M, int N, int K)
{
    __shared__ _Float16 As [GBM][GBK + GPAD];
    __shared__ _Float16 BsT[GBN][GBK + GPAD];   // transposed: [n][k]

    const float* Bw = (blockIdx.z == 0) ? B0 : (blockIdx.z == 1 ? B1 : B2);
    void* Cout      = (blockIdx.z == 0) ? C0 : (blockIdx.z == 1 ? C1 : C2);

    const int tid    = threadIdx.x;
    const int lane   = tid & 31;
    const int wid    = tid >> 5;
    const int bm     = blockIdx.y * GBM;
    const int bn     = blockIdx.x * GBN;
    const int wave_m = (wid >> 2) * 64;   // 2 waves over M
    const int wave_n = (wid & 3)  * 32;   // 4 waves over N

    // WMMA lane decomposition (per ISA 16x16x32 f16 layouts)
    const int row   = lane & 15;          // A-frag row
    const int klo   = (lane >> 4) * 8;    // A-frag K base (0 / 8)
    const int klo2  = (lane >> 4) * 16;   // unused (0 / 16); B-frag also uses klo
    const int mbase = (lane >> 4) * 8;    // C rows 0-7 / 8-15
    const int ncol  = lane & 15;          // B/C column

    v8f zero = {};
    v8f acc[4][2];
    for (int mt = 0; mt < 4; ++mt)
        for (int nt = 0; nt < 2; ++nt) acc[mt][nt] = zero;

    for (int k0 = 0; k0 < K; k0 += GBK) {
        // ---- prefetch next K-tile into cache while this stage computes ----
        if (k0 + GBK < K) {
            {   // A: 128 rows x GBK elems
                int r = tid & 127;
                int half = tid >> 7;
                __builtin_prefetch(A + (size_t)(bm + r) * K + (k0 + GBK)
                                     + half * (GBK / 2), 0, 3);
            }
            {   // B: GBK rows x 128 cols
                int kr = tid >> 2;
                int qd = tid & 3;
                __builtin_prefetch(Bw + (size_t)(k0 + GBK + kr) * N + bn
                                      + qd * 32, 0, 3);
            }
        }
        __syncthreads();
        // ---- stage A tile (convert to f16 if fp32) ----
        if constexpr (sizeof(AT) == 4) {
            for (int i = 0; i < 8; ++i) {             // 2048 float4 loads
                int vi = tid + i * 256;
                int r  = vi >> 4;                     // 16 vec4 per 64-col row
                int c  = (vi & 15) * 4;
                float4 a4 = *(const float4*)(A + (size_t)(bm + r) * K + k0 + c);
                H4 p;
                p.h[0] = (_Float16)a4.x;
                p.h[1] = (_Float16)a4.y;
                p.h[2] = (_Float16)a4.z;
                p.h[3] = (_Float16)a4.w;
                *(uint2*)(&As[r][c]) = p.u;           // one ds_store_b64
            }
        } else {
            for (int i = 0; i < 4; ++i) {             // 1024 16B copies
                int vi = tid + i * 256;
                int r  = vi >> 3;                     // 8 vec8 per 64-col row
                int c  = (vi & 7) * 8;
                *(uint4*)(&As[r][c]) =
                    *(const uint4*)(A + (size_t)(bm + r) * K + k0 + c);
            }
        }
        // ---- stage B tile transposed ----
        for (int i = 0; i < 8; ++i) {                 // 2048 float4 loads
            int vi = tid + i * 256;
            int kr = vi >> 5;                         // 32 vec4 per 128-col row
            int c  = (vi & 31) * 4;
            float4 b4 = *(const float4*)(Bw + (size_t)(k0 + kr) * N + bn + c);
            BsT[c + 0][kr] = (_Float16)b4.x;
            BsT[c + 1][kr] = (_Float16)b4.y;
            BsT[c + 2][kr] = (_Float16)b4.z;
            BsT[c + 3][kr] = (_Float16)b4.w;
        }
        __syncthreads();

        // ---- 2 WMMA k-steps per stage ----
        for (int ks = 0; ks < 2; ++ks) {
            FragAB af[4], bf[2];
            for (int mt = 0; mt < 4; ++mt) {
                const _Float16* ap = &As[wave_m + mt * 16 + row][ks * 32 + klo];
                af[mt].q[0] = *(const uint4*)(ap);        // K = klo..klo+7
                af[mt].q[1] = *(const uint4*)(ap + 16);   // K = 16+klo..+7
            }
            for (int nt = 0; nt < 2; ++nt) {
                const _Float16* bp = &BsT[wave_n + nt * 16 + ncol][ks * 32 + klo];
                bf[nt].q[0] = *(const uint4*)(bp);        // K = klo..klo+7
                bf[nt].q[1] = *(const uint4*)(bp + 16);   // K = 16+klo..+7
            }
            for (int mt = 0; mt < 4; ++mt)
                for (int nt = 0; nt < 2; ++nt)
                    acc[mt][nt] = wmma_f16(af[mt].v, bf[nt].v, acc[mt][nt]);
        }
    }

    // ---- epilogue ----
    for (int mt = 0; mt < 4; ++mt)
        for (int nt = 0; nt < 2; ++nt)
            for (int j = 0; j < 8; ++j) {
                int r = bm + wave_m + mt * 16 + mbase + j;
                int c = bn + wave_n + nt * 16 + ncol;
                float val = acc[mt][nt][j];
                if constexpr (OUT_MODE == 0) {
                    ((float*)Cout)[(size_t)r * N + c] = val;
                } else {
                    int b = r >> 10, s = r & (S - 1);
                    int h = c >> 6,  d = c & (DK - 1);
                    ((_Float16*)Cout)[(((size_t)(b * H + h)) * S + s) * DK + d]
                        = (_Float16)val;
                }
            }
}

// ---------------------------------------------------------------------------
// Fused attention: softmax(Q K^T + bias) V, flash-style over key chunks.
// Grid: (S/128, H, B). 256 threads = 8 waves; wave owns 16 q-rows.
// q/k/v f16 in [B,H,S,DK]; ctx f16 out in [B,S,H*DK].
// ---------------------------------------------------------------------------
__global__ __launch_bounds__(256) void attn_kernel(
    const _Float16* __restrict__ qh, const _Float16* __restrict__ kh,
    const _Float16* __restrict__ vh, const float* __restrict__ rel_emb,
    _Float16* __restrict__ ctx)
{
    __shared__ _Float16 Ks[128][DK + 8];        // [key][d], stride 144B
    __shared__ _Float16 VsT[DK][128 + 8];       // [d][key], stride 272B
    __shared__ _Float16 Pbuf[8][16][32 + 8];    // per-wave P slice, stride 80B
    __shared__ float bias_t[2 * S];             // bias vs (key - q)

    const int tid  = threadIdx.x;
    const int lane = tid & 31;
    const int wid  = tid >> 5;
    const int q0   = blockIdx.x * 128;
    const int h    = blockIdx.y;
    const int b    = blockIdx.z;

    // --- per-block bias table: bias_t[(key-q) + (S-1)] = rel_emb[bucket][h]
    for (int r = tid; r < 2 * S - 1; r += 256) {
        int rel = r - (S - 1);                  // memory - query
        int n   = -rel;
        int ret = (n < 0) ? (NUM_BUCKETS / 2) : 0;
        int na  = (n < 0) ? -n : n;
        int bucket;
        if (na < 8) {
            bucket = ret + na;
        } else {
            // 8 + trunc(log(na/8)/log(16) * 8), capped at 15
            int v = 8 + (int)(logf((float)na * 0.125f) * 2.8853900817779268f);
            bucket = ret + (v < 15 ? v : 15);
        }
        bias_t[r] = rel_emb[bucket * H + h];
    }

    const size_t head_off = ((size_t)(b * H + h)) * S * DK;
    const _Float16* Qp = qh + head_off;
    const _Float16* Kp = kh + head_off;
    const _Float16* Vp = vh + head_off;

    const int row   = lane & 15;
    const int klo   = (lane >> 4) * 8;
    const int klo2  = (lane >> 4) * 16;
    const int mbase = (lane >> 4) * 8;
    const int ncol  = lane & 15;
    const int qrow_base = q0 + wid * 16;

    // --- Q fragments resident in registers (DK=64 -> 2 k-steps)
    FragAB qf[2];
    {
        const _Float16* qp = Qp + (size_t)(qrow_base + row) * DK;
        for (int ks = 0; ks < 2; ++ks) {
            qf[ks].q[0] = *(const uint4*)(qp + ks * 32 + klo);
            qf[ks].q[1] = *(const uint4*)(qp + ks * 32 + 16 + klo);
        }
    }

    v8f zero = {};
    float m_run[8], l_run[8];
    v8f o_acc[4];
    for (int j = 0; j < 8; ++j) { m_run[j] = -1e30f; l_run[j] = 0.0f; }
    for (int dt = 0; dt < 4; ++dt) o_acc[dt] = zero;

    for (int kc = 0; kc < S; kc += 128) {
        // ---- prefetch next key/value chunk (each row = one 128B line) ----
        if (kc + 128 < S) {
            int key  = tid & 127;
            const _Float16* src = (tid < 128) ? Kp : Vp;
            __builtin_prefetch(src + (size_t)(kc + 128 + key) * DK, 0, 3);
        }
        __syncthreads();
        // ---- stage K chunk [key][d] and V chunk transposed [d][key] ----
        for (int i = 0; i < 4; ++i) {
            int vi  = tid + i * 256;
            int key = vi >> 3;
            int dv  = (vi & 7) * 8;
            *(uint4*)(&Ks[key][dv]) =
                *(const uint4*)(Kp + (size_t)(kc + key) * DK + dv);
            uint4 vd = *(const uint4*)(Vp + (size_t)(kc + key) * DK + dv);
            const _Float16* v8 = (const _Float16*)&vd;
            for (int j = 0; j < 8; ++j) VsT[dv + j][key] = v8[j];
        }
        __syncthreads();

        // ---- scores: 8 tiles of 16x16, K-dim = DK in 2 steps ----
        v8f sc[8];
        for (int nt = 0; nt < 8; ++nt) sc[nt] = zero;
        for (int ks = 0; ks < 2; ++ks) {
            for (int nt = 0; nt < 8; ++nt) {
                FragAB bf;
                const _Float16* kp = &Ks[nt * 16 + ncol][ks * 32 + klo];
                bf.q[0] = *(const uint4*)(kp);
                bf.q[1] = *(const uint4*)(kp + 16);
                sc[nt] = wmma_f16(qf[ks].v, bf.v, sc[nt]);
            }
        }
        // ---- add relative-position bias ----
        for (int nt = 0; nt < 8; ++nt)
            for (int j = 0; j < 8; ++j) {
                int key = kc + nt * 16 + ncol;
                int qr  = qrow_base + mbase + j;
                sc[nt][j] += bias_t[key - qr + (S - 1)];
            }
        // ---- streaming softmax update ----
        float mnew[8], scale[8];
        for (int j = 0; j < 8; ++j) {
            float v = sc[0][j];
            for (int nt = 1; nt < 8; ++nt) v = fmaxf(v, sc[nt][j]);
            v = fmaxf(v, __shfl_xor(v, 1));
            v = fmaxf(v, __shfl_xor(v, 2));
            v = fmaxf(v, __shfl_xor(v, 4));
            v = fmaxf(v, __shfl_xor(v, 8));
            mnew[j]  = fmaxf(m_run[j], v);
            scale[j] = expf(m_run[j] - mnew[j]);
            m_run[j] = mnew[j];
        }
        float rs[8];
        for (int j = 0; j < 8; ++j) rs[j] = 0.0f;
        for (int nt = 0; nt < 8; ++nt)
            for (int j = 0; j < 8; ++j) {
                float p = expf(sc[nt][j] - mnew[j]);
                sc[nt][j] = p;
                rs[j] += p;
            }
        for (int j = 0; j < 8; ++j) {
            float v = rs[j];
            v += __shfl_xor(v, 1);
            v += __shfl_xor(v, 2);
            v += __shfl_xor(v, 4);
            v += __shfl_xor(v, 8);
            l_run[j] = l_run[j] * scale[j] + v;
        }
        for (int dt = 0; dt < 4; ++dt)
            for (int j = 0; j < 8; ++j) o_acc[dt][j] *= scale[j];

        // ---- P @ V : 4 k-steps of 32 keys each ----
        for (int kk = 0; kk < 4; ++kk) {
            // C-layout -> A-layout via per-wave LDS slice (in-order per wave)
            for (int t = 0; t < 2; ++t)
                for (int j = 0; j < 8; ++j)
                    Pbuf[wid][mbase + j][t * 16 + ncol] =
                        (_Float16)sc[kk * 2 + t][j];
            asm volatile("" ::: "memory");  // keep store->load order
            FragAB pf;
            const _Float16* pp = &Pbuf[wid][row][klo];
            pf.q[0] = *(const uint4*)(pp);
            pf.q[1] = *(const uint4*)(pp + 16);
            for (int dt = 0; dt < 4; ++dt) {
                FragAB vf;
                const _Float16* vp = &VsT[dt * 16 + ncol][kk * 32 + klo];
                vf.q[0] = *(const uint4*)(vp);
                vf.q[1] = *(const uint4*)(vp + 16);
                o_acc[dt] = wmma_f16(pf.v, vf.v, o_acc[dt]);
            }
        }
    }

    // ---- normalize and write ctx [B,S,H*DK] as f16 ----
    for (int j = 0; j < 8; ++j) {
        float inv  = 1.0f / l_run[j];
        int   srow = qrow_base + mbase + j;
        _Float16* cp = ctx + ((size_t)(b * S + srow)) * INNER + h * DK;
        for (int dt = 0; dt < 4; ++dt)
            cp[dt * 16 + ncol] = (_Float16)(o_acc[dt][j] * inv);
    }
}

// ---------------------------------------------------------------------------
// Launch
// ---------------------------------------------------------------------------
extern "C" void kernel_launch(void* const* d_in, const int* in_sizes, int n_in,
                              void* d_out, int out_size, void* d_ws, size_t ws_size,
                              hipStream_t stream)
{
    const float* x   = (const float*)d_in[0];
    const float* Wq  = (const float*)d_in[1];
    const float* Wk  = (const float*)d_in[2];
    const float* Wv  = (const float*)d_in[3];
    const float* Wo  = (const float*)d_in[4];
    const float* rel = (const float*)d_in[5];

    const size_t HEAD_ELEMS = (size_t)BB * H * S * DK;  // 4M halves each
    _Float16* qh  = (_Float16*)d_ws;
    _Float16* khp = qh  + HEAD_ELEMS;
    _Float16* vhp = khp + HEAD_ELEMS;
    _Float16* ctx = vhp + HEAD_ELEMS;                   // [B,S,INNER] f16

    dim3 gblk(256);

    // Q, K, V projections fused into one launch (gridDim.z = 3):
    // 768 blocks in flight, shared x tiles hit in L2.
    dim3 qkv_grid(INNER / GBN, M_ALL / GBM, 3);         // (8, 32, 3)
    gemm_kernel<float, 1><<<qkv_grid, gblk, 0, stream>>>(
        x, Wq, Wk, Wv, (void*)qh, (void*)khp, (void*)vhp, M_ALL, INNER, D);

    // fused biased attention -> ctx f16 [B,S,INNER]
    dim3 agrid(S / 128, H, BB);                         // (8, 16, 4)
    attn_kernel<<<agrid, gblk, 0, stream>>>(qh, khp, vhp, rel, ctx);

    // output projection -> fp32 d_out [B,S,D]
    dim3 ogrid(D / GBN, M_ALL / GBM, 1);                // (8, 32, 1)
    gemm_kernel<_Float16, 0><<<ogrid, gblk, 0, stream>>>(
        ctx, Wo, Wo, Wo, d_out, d_out, d_out, M_ALL, D, INNER);
}
